// MoE_33045478375625
// MI455X (gfx1250) — compile-verified, matrix operand order checked by reading, not run on hardware
//
#include <hip/hip_runtime.h>

#define NE 8          // experts
#define TKK 2         // top-k
#define HD 2048       // hidden
#define ID 1024       // intermediate
#define NT 4096       // tokens

#define TM 128        // tile rows (tokens)
#define TN 128        // tile cols (features)
#define BK 32         // K step = one bf16 WMMA K
#define LDK 40        // padded LDS K stride (halfwords) -> 80B rows, 16B aligned

typedef __bf16 bf16_t;
typedef __attribute__((ext_vector_type(16))) __bf16 v16bf;
typedef __attribute__((ext_vector_type(4)))  __bf16 v4bf;
typedef __attribute__((ext_vector_type(8)))  float  v8f;
typedef __attribute__((ext_vector_type(4)))  unsigned int u32x4;

union Frag16 { v16bf v; u32x4 q[2]; };

// ---------------------------------------------------------------- init
__global__ __launch_bounds__(256) void zero_out_kernel(float4* __restrict__ out) {
    size_t i = (size_t)blockIdx.x * blockDim.x + threadIdx.x;
    out[i] = make_float4(0.f, 0.f, 0.f, 0.f);   // grid sized exactly: NT*HD/4 elems
}

__global__ void zero_counts_kernel(int* __restrict__ counts) {
    if (threadIdx.x < 16) counts[threadIdx.x] = 0;
}

// ---------------------------------------------------------------- router
// one wave32 per token: 8 dots over H, softmax, top-2, compaction lists
__global__ __launch_bounds__(256) void router_kernel(
    const float* __restrict__ x, const float* __restrict__ rw,
    int* __restrict__ counts, int* __restrict__ pair_list,
    float* __restrict__ gate_ws)
{
    int lane = threadIdx.x & 31;
    int t = blockIdx.x * 8 + (threadIdx.x >> 5);
    if (t >= NT) return;
    const float* xr = x + (size_t)t * HD;

    float acc[NE];
#pragma unroll
    for (int e = 0; e < NE; ++e) acc[e] = 0.f;

    for (int h = lane * 4; h < HD; h += 32 * 4) {
        float4 xv = *(const float4*)(xr + h);
#pragma unroll
        for (int e = 0; e < NE; ++e) {
            float4 wv = *(const float4*)(rw + (size_t)e * HD + h);
            acc[e] += xv.x * wv.x + xv.y * wv.y + xv.z * wv.z + xv.w * wv.w;
        }
    }
#pragma unroll
    for (int off = 16; off >= 1; off >>= 1)
#pragma unroll
        for (int e = 0; e < NE; ++e) acc[e] += __shfl_xor(acc[e], off, 32);

    if (lane == 0) {
        float mx = acc[0];
#pragma unroll
        for (int e = 1; e < NE; ++e) mx = fmaxf(mx, acc[e]);
        float p[NE], s = 0.f;
#pragma unroll
        for (int e = 0; e < NE; ++e) { p[e] = __expf(acc[e] - mx); s += p[e]; }
        float inv = 1.f / s;
        int i0 = 0;
#pragma unroll
        for (int e = 1; e < NE; ++e) if (acc[e] > acc[i0]) i0 = e;      // ties -> lowest idx
        int i1 = (i0 == 0) ? 1 : 0;
#pragma unroll
        for (int e = 0; e < NE; ++e) if (e != i0 && acc[e] > acc[i1]) i1 = e;

        gate_ws[t * TKK + 0] = p[i0] * inv;
        gate_ws[t * TKK + 1] = p[i1] * inv;
        int pos0 = atomicAdd(&counts[i0], 1);
        pair_list[i0 * NT + pos0] = t * TKK + 0;
        int pos1 = atomicAdd(&counts[i1], 1);
        pair_list[i1 * NT + pos1] = t * TKK + 1;
    }
}

// ---------------------------------------------------------------- up proj + swiglu
// C-tile 128x128, 8 waves, each wave 32x64 (2x4 frags of 16x16).
// Double-buffered LDS, one barrier per K step; unconditional clamped gather loads
// (padded rows only pollute C rows that are never stored).
__global__ __launch_bounds__(256) void moe_up_kernel(
    const float* __restrict__ x, const float* __restrict__ w1,
    const int* __restrict__ counts, const int* __restrict__ pair_list,
    bf16_t* __restrict__ hbuf)
{
    int e = blockIdx.z;
    int cnt = counts[e];
    int m_base = blockIdx.y * TM;
    if (m_base >= cnt) return;
    int n_base = blockIdx.x * TN;         // over 2*ID features

    __shared__ __attribute__((aligned(16))) bf16_t As[2][TM][LDK];
    __shared__ __attribute__((aligned(16))) bf16_t Bs[2][TN][LDK];
    __shared__ int rowslot[TM];

    int tid  = threadIdx.x;
    int lane = tid & 31;
    int wid  = tid >> 5;
    int wm   = (wid & 3) * 32;
    int wn   = (wid >> 2) * 64;

    if (tid < TM) {
        int m = m_base + tid;
        rowslot[tid] = (m < cnt) ? pair_list[e * NT + m] : -1;
    }
    __syncthreads();

    // loop-invariant per-thread load pointers (4 float4 slots per thread)
    const float* w1e = w1 + (size_t)e * (2 * ID) * HD;
    const float* aptr[4];
    const float* bptr[4];
#pragma unroll
    for (int i = 0; i < 4; ++i) {
        int s  = tid + i * 256;
        int r  = s >> 3;                          // 8 float4 per 32-wide row
        int kq = (s & 7) << 2;
        int sl  = rowslot[r];
        int tok = (sl < 0) ? 0 : (sl >> 1);       // clamp: no branchy loads
        aptr[i] = x   + (size_t)tok * HD + kq;
        bptr[i] = w1e + (size_t)(n_base + r) * HD + kq;
    }

    v8f c[2][4];
#pragma unroll
    for (int mi = 0; mi < 2; ++mi)
#pragma unroll
        for (int ni = 0; ni < 4; ++ni) c[mi][ni] = (v8f){0,0,0,0,0,0,0,0};

    float4 av[4], bv[4];
    // prologue: stage k=0
#pragma unroll
    for (int i = 0; i < 4; ++i) { av[i] = *(const float4*)aptr[i]; bv[i] = *(const float4*)bptr[i]; }
#pragma unroll
    for (int i = 0; i < 4; ++i) {
        int s = tid + i * 256; int r = s >> 3; int kq = (s & 7) << 2;
        *(v4bf*)&As[0][r][kq] = (v4bf){(bf16_t)av[i].x, (bf16_t)av[i].y, (bf16_t)av[i].z, (bf16_t)av[i].w};
        *(v4bf*)&Bs[0][r][kq] = (v4bf){(bf16_t)bv[i].x, (bf16_t)bv[i].y, (bf16_t)bv[i].z, (bf16_t)bv[i].w};
    }
    __syncthreads();

    const int nk = HD / BK;
    int koff = (lane < 16) ? 0 : 8;               // ISA bf16 A/B K pattern
    int r16  = lane & 15;

    for (int k = 0; k < nk; ++k) {
        int  buf  = k & 1;
        bool more = (k + 1 < nk);
        if (more) {                                // issue next-tile loads before WMMAs
            int kn = (k + 1) * BK;
#pragma unroll
            for (int i = 0; i < 4; ++i) {
                av[i] = *(const float4*)(aptr[i] + kn);
                bv[i] = *(const float4*)(bptr[i] + kn);
            }
        }

        Frag16 a[2], b[4];
#pragma unroll
        for (int mi = 0; mi < 2; ++mi) {
            const bf16_t* p = &As[buf][wm + mi * 16 + r16][koff];
            a[mi].q[0] = *(const u32x4*)p;
            a[mi].q[1] = *(const u32x4*)(p + 16);
        }
#pragma unroll
        for (int ni = 0; ni < 4; ++ni) {
            const bf16_t* p = &Bs[buf][wn + ni * 16 + r16][koff];
            b[ni].q[0] = *(const u32x4*)p;
            b[ni].q[1] = *(const u32x4*)(p + 16);
        }
#pragma unroll
        for (int mi = 0; mi < 2; ++mi)
#pragma unroll
            for (int ni = 0; ni < 4; ++ni)
                c[mi][ni] = __builtin_amdgcn_wmma_f32_16x16x32_bf16(
                    false, a[mi].v, false, b[ni].v, (short)0, c[mi][ni], false, false);

        if (more) {
#pragma unroll
            for (int i = 0; i < 4; ++i) {
                int s = tid + i * 256; int r = s >> 3; int kq = (s & 7) << 2;
                *(v4bf*)&As[buf ^ 1][r][kq] =
                    (v4bf){(bf16_t)av[i].x, (bf16_t)av[i].y, (bf16_t)av[i].z, (bf16_t)av[i].w};
                *(v4bf*)&Bs[buf ^ 1][r][kq] =
                    (v4bf){(bf16_t)bv[i].x, (bf16_t)bv[i].y, (bf16_t)bv[i].z, (bf16_t)bv[i].w};
            }
        }
        __syncthreads();
    }

    // SwiGLU: even feature = gate, odd = up; adjacent N cols sit in adjacent lanes
    int lane16 = lane & 15;
    int mhalf  = (lane >> 4) << 3;
#pragma unroll
    for (int mi = 0; mi < 2; ++mi) {
#pragma unroll
        for (int j = 0; j < 8; ++j) {
            int lr = wm + mi * 16 + mhalf + j;
            int m  = m_base + lr;
            bool rowok = (m < cnt);
            int slot = rowslot[lr];
#pragma unroll
            for (int ni = 0; ni < 4; ++ni) {
                float v = c[mi][ni][j];
                float partner = __shfl_xor(v, 1, 32);
                if (rowok && ((lane16 & 1) == 0)) {
                    float g = v, u = partner;
                    float hv = u * g / (1.f + __expf(-g));
                    int ncol = n_base + wn + ni * 16 + lane16;   // even
                    hbuf[(size_t)slot * ID + (ncol >> 1)] = (bf16_t)hv;
                }
            }
        }
    }
}

// ---------------------------------------------------------------- down proj + gate + scatter
__global__ __launch_bounds__(256) void moe_down_kernel(
    const bf16_t* __restrict__ hbuf, const float* __restrict__ w2,
    const int* __restrict__ counts, const int* __restrict__ pair_list,
    const float* __restrict__ gate_ws, float* __restrict__ out)
{
    int e = blockIdx.z;
    int cnt = counts[e];
    int m_base = blockIdx.y * TM;
    if (m_base >= cnt) return;
    int n_base = blockIdx.x * TN;         // over HD outputs

    __shared__ __attribute__((aligned(16))) bf16_t As[2][TM][LDK];
    __shared__ __attribute__((aligned(16))) bf16_t Bs[2][TN][LDK];
    __shared__ int rowslot[TM];

    int tid  = threadIdx.x;
    int lane = tid & 31;
    int wid  = tid >> 5;
    int wm   = (wid & 3) * 32;
    int wn   = (wid >> 2) * 64;

    if (tid < TM) {
        int m = m_base + tid;
        rowslot[tid] = (m < cnt) ? pair_list[e * NT + m] : -1;
    }
    __syncthreads();

    const float* w2e = w2 + (size_t)e * HD * ID;
    const bf16_t* aptr[2];                        // 2x (8 bf16) slots per thread
    const float*  bptr[4];
#pragma unroll
    for (int i = 0; i < 2; ++i) {
        int s  = tid + i * 256;
        int r  = s >> 2;                          // 4x(8 bf16) per 32-wide row
        int kq = (s & 3) << 3;
        int sl = rowslot[r];
        int row = (sl < 0) ? 0 : sl;              // clamp: padded rows never stored
        aptr[i] = hbuf + (size_t)row * ID + kq;
    }
#pragma unroll
    for (int i = 0; i < 4; ++i) {
        int s  = tid + i * 256;
        int r  = s >> 3;
        int kq = (s & 7) << 2;
        bptr[i] = w2e + (size_t)(n_base + r) * ID + kq;
    }

    v8f c[2][4];
#pragma unroll
    for (int mi = 0; mi < 2; ++mi)
#pragma unroll
        for (int ni = 0; ni < 4; ++ni) c[mi][ni] = (v8f){0,0,0,0,0,0,0,0};

    u32x4 av[2];
    float4 bv[4];
    // prologue: stage k=0
#pragma unroll
    for (int i = 0; i < 2; ++i) av[i] = *(const u32x4*)aptr[i];
#pragma unroll
    for (int i = 0; i < 4; ++i) bv[i] = *(const float4*)bptr[i];
#pragma unroll
    for (int i = 0; i < 2; ++i) {
        int s = tid + i * 256; int r = s >> 2; int kq = (s & 3) << 3;
        *(u32x4*)&As[0][r][kq] = av[i];
    }
#pragma unroll
    for (int i = 0; i < 4; ++i) {
        int s = tid + i * 256; int r = s >> 3; int kq = (s & 7) << 2;
        *(v4bf*)&Bs[0][r][kq] = (v4bf){(bf16_t)bv[i].x, (bf16_t)bv[i].y, (bf16_t)bv[i].z, (bf16_t)bv[i].w};
    }
    __syncthreads();

    const int nk = ID / BK;
    int koff = (lane < 16) ? 0 : 8;
    int r16  = lane & 15;

    for (int k = 0; k < nk; ++k) {
        int  buf  = k & 1;
        bool more = (k + 1 < nk);
        if (more) {
            int kn = (k + 1) * BK;
#pragma unroll
            for (int i = 0; i < 2; ++i) av[i] = *(const u32x4*)(aptr[i] + kn);
#pragma unroll
            for (int i = 0; i < 4; ++i) bv[i] = *(const float4*)(bptr[i] + kn);
        }

        Frag16 a[2], b[4];
#pragma unroll
        for (int mi = 0; mi < 2; ++mi) {
            const bf16_t* p = &As[buf][wm + mi * 16 + r16][koff];
            a[mi].q[0] = *(const u32x4*)p;
            a[mi].q[1] = *(const u32x4*)(p + 16);
        }
#pragma unroll
        for (int ni = 0; ni < 4; ++ni) {
            const bf16_t* p = &Bs[buf][wn + ni * 16 + r16][koff];
            b[ni].q[0] = *(const u32x4*)p;
            b[ni].q[1] = *(const u32x4*)(p + 16);
        }
#pragma unroll
        for (int mi = 0; mi < 2; ++mi)
#pragma unroll
            for (int ni = 0; ni < 4; ++ni)
                c[mi][ni] = __builtin_amdgcn_wmma_f32_16x16x32_bf16(
                    false, a[mi].v, false, b[ni].v, (short)0, c[mi][ni], false, false);

        if (more) {
#pragma unroll
            for (int i = 0; i < 2; ++i) {
                int s = tid + i * 256; int r = s >> 2; int kq = (s & 3) << 3;
                *(u32x4*)&As[buf ^ 1][r][kq] = av[i];
            }
#pragma unroll
            for (int i = 0; i < 4; ++i) {
                int s = tid + i * 256; int r = s >> 3; int kq = (s & 7) << 2;
                *(v4bf*)&Bs[buf ^ 1][r][kq] =
                    (v4bf){(bf16_t)bv[i].x, (bf16_t)bv[i].y, (bf16_t)bv[i].z, (bf16_t)bv[i].w};
            }
        }
        __syncthreads();
    }

    int lane16 = lane & 15;
    int mhalf  = (lane >> 4) << 3;
#pragma unroll
    for (int mi = 0; mi < 2; ++mi) {
#pragma unroll
        for (int j = 0; j < 8; ++j) {
            int lr = wm + mi * 16 + mhalf + j;
            int m  = m_base + lr;
            if (m < cnt) {
                int slot   = rowslot[lr];
                int token  = slot >> 1;
                float gate = gate_ws[slot];
                float* orow = out + (size_t)token * HD;
#pragma unroll
                for (int ni = 0; ni < 4; ++ni) {
                    int ncol = n_base + wn + ni * 16 + lane16;
                    atomicAdd(orow + ncol, gate * c[mi][ni][j]);  // exactly 2 adds/elem -> deterministic
                }
            }
        }
    }
}

// ---------------------------------------------------------------- launch
extern "C" void kernel_launch(void* const* d_in, const int* in_sizes, int n_in,
                              void* d_out, int out_size, void* d_ws, size_t ws_size,
                              hipStream_t stream) {
    (void)in_sizes; (void)n_in; (void)out_size; (void)ws_size;
    const float* x  = (const float*)d_in[0];
    const float* rw = (const float*)d_in[1];
    const float* w1 = (const float*)d_in[2];
    const float* w2 = (const float*)d_in[3];
    float* out = (float*)d_out;

    // workspace layout (~17 MB): counts | pair_list | gates | h(bf16)
    char* ws = (char*)d_ws;
    int*    counts    = (int*)ws;                                   // 256 B pad
    int*    pair_list = (int*)(ws + 256);                           // NE*NT*4 = 128 KB
    float*  gate_ws   = (float*)(ws + 256 + NE * NT * 4);           // NT*TKK*4 = 32 KB
    bf16_t* hbuf      = (bf16_t*)(ws + 256 + NE * NT * 4 + NT * TKK * 4); // NT*TKK*ID*2

    zero_counts_kernel<<<1, 32, 0, stream>>>(counts);
    zero_out_kernel<<<(NT * HD / 4) / 256, 256, 0, stream>>>((float4*)out);
    router_kernel<<<NT / 8, 256, 0, stream>>>(x, rw, counts, pair_list, gate_ws);

    dim3 gup(2 * ID / TN, NT / TM, NE);   // (16, 32, 8)
    moe_up_kernel<<<gup, 256, 0, stream>>>(x, w1, counts, pair_list, hbuf);

    dim3 gdn(HD / TN, NT / TM, NE);       // (16, 32, 8)
    moe_down_kernel<<<gdn, 256, 0, stream>>>(hbuf, w2, counts, pair_list, gate_ws, out);
}
